// Flow_16063177687349
// MI455X (gfx1250) — matrix operand order repaired and checked by reading, not compiled, not run on hardware
//
#include <hip/hip_runtime.h>
#include <hip/hip_bf16.h>

typedef __bf16 bf16_t;
typedef __attribute__((ext_vector_type(8)))  __bf16 v8bf;
typedef __attribute__((ext_vector_type(16))) __bf16 v16bf;
typedef __attribute__((ext_vector_type(8)))  float  v8f;

#define T_LEN 1024
#define BATCH 64
#define DDIM  256
#define MDIM  128
#define PDIM  64

#define XKP 264   // bf16 k-pitch, x tile rows in LDS (132 dw -> 4-bank lane step)
#define VKP 260   // f32 pitch of V rows in LDS (breaks 8-row bank aliasing)
#define BKP 136   // bf16 k-pitch, column-major Bm in LDS
#define CKP 264   // bf16 k-pitch, column-major C in LDS

// Dopri5 tableau
static constexpr float AT_[6][5] = {
  {0.f, 0.f, 0.f, 0.f, 0.f},
  {0.2f, 0.f, 0.f, 0.f, 0.f},
  {3.f/40.f, 9.f/40.f, 0.f, 0.f, 0.f},
  {44.f/45.f, -56.f/15.f, 32.f/9.f, 0.f, 0.f},
  {19372.f/6561.f, -25360.f/2187.f, 64448.f/6561.f, -212.f/729.f, 0.f},
  {9017.f/3168.f, -355.f/33.f, 46732.f/5247.f, 49.f/176.f, -5103.f/18656.f}};
static constexpr float BT_[6] = {35.f/384.f, 0.f, 500.f/1113.f, 125.f/192.f,
                                 -2187.f/6784.f, 11.f/84.f};

static __device__ __forceinline__ v16bf cat16(v8bf lo, v8bf hi) {
  return __builtin_shufflevector(lo, hi, 0,1,2,3,4,5,6,7,8,9,10,11,12,13,14,15);
}
static __device__ __forceinline__ v16bf frag_lds(const bf16_t* p0, const bf16_t* p1) {
  v8bf lo = *(const v8bf*)p0;
  v8bf hi = *(const v8bf*)p1;
  return cat16(lo, hi);
}
static __device__ __forceinline__ v8bf cvt8(const float* p) {
  const float4* q = (const float4*)p;
  float4 a = q[0], b = q[1];
  v8bf r;
  r[0]=(bf16_t)a.x; r[1]=(bf16_t)a.y; r[2]=(bf16_t)a.z; r[3]=(bf16_t)a.w;
  r[4]=(bf16_t)b.x; r[5]=(bf16_t)b.y; r[6]=(bf16_t)b.z; r[7]=(bf16_t)b.w;
  return r;
}
static __device__ __forceinline__ v8f wmma_bf16(v16bf a, v16bf b, v8f c) {
  return __builtin_amdgcn_wmma_f32_16x16x32_bf16(false, a, false, b, (short)0, c,
                                                 false, false);
}

// Async prefetch of one V time-slice (16 batch rows x 256 f32 = 16KB) into an
// LDS ring slot via CDNA5 async-to-LDS ops (tracked by ASYNCcnt).
// 512 threads -> 2 x b128 chunks per thread.
static __device__ __forceinline__ void async_v(const float* Vg, int tp, int bBase,
                                               unsigned lds_byte, int tid) {
  unsigned long long g =
      (unsigned long long)(size_t)(const void*)(Vg + ((size_t)tp * BATCH + bBase) * DDIM);
#pragma unroll
  for (int i = 0; i < 2; ++i) {
    int c = tid + (i << 9);          // 0..1023 chunks of 16B
    int row = c >> 6;                // 0..15
    int off = (c & 63) << 4;         // byte offset within 1KB source row
    unsigned lds  = lds_byte + (unsigned)(row * (VKP * 4) + off);
    unsigned goff = (unsigned)(row * (DDIM * 4) + off);
    asm volatile("global_load_async_to_lds_b128 %0, %1, %2"
                 :: "v"(lds), "v"(goff), "s"(g)
                 : "memory");
  }
}

// ---------------------------------------------------------------------------
// Kernel 1: V[t][b][:] = u[b][t][:] @ Bm   (M=65536, N=256, K=128)
// ---------------------------------------------------------------------------
__global__ void __launch_bounds__(256)
v_gemm_kernel(const float* __restrict__ u, const float* __restrict__ Bm,
              float* __restrict__ V) {
  extern __shared__ char smem[];
  bf16_t* Bcm = (bf16_t*)smem;   // [256][BKP] column-major bf16
  const int tid = threadIdx.x;
  for (int i = tid; i < MDIM * DDIM; i += 256) {
    int k = i >> 8, n = i & 255;
    Bcm[(size_t)n * BKP + k] = (bf16_t)Bm[i];
  }
  __syncthreads();
  const int lane = tid & 31, wave = tid >> 5;
  const int h = lane >> 4, nn = lane & 15, mA = lane & 15;
  for (int task = blockIdx.x * 8 + wave; task < (T_LEN * BATCH / 16) * 4; task += 512) {
    int mt = task >> 2, n0 = (task & 3) * 64;
    int b = (mt & 3) * 16 + mA;     // flat row R = t*64+b  ->  b = R%64
    int t = mt >> 2;
    const float* urow = u + ((size_t)b * T_LEN + t) * MDIM;
    v8f acc[4] = {};
#pragma unroll
    for (int kt = 0; kt < 4; ++kt) {
      int kb = kt * 32;
      v16bf af = cat16(cvt8(urow + kb + h * 8), cvt8(urow + kb + 16 + h * 8));
#pragma unroll
      for (int nt = 0; nt < 4; ++nt) {
        const bf16_t* bc = Bcm + (size_t)(n0 + nt * 16 + nn) * BKP + kb + h * 8;
        acc[nt] = wmma_bf16(af, frag_lds(bc, bc + 16), acc[nt]);
      }
    }
#pragma unroll
    for (int nt = 0; nt < 4; ++nt)
#pragma unroll
      for (int r = 0; r < 8; ++r)
        V[(size_t)(mt * 16 + r + 8 * h) * DDIM + n0 + nt * 16 + nn] = acc[nt][r];
  }
}

// ---------------------------------------------------------------------------
// Kernel 2: sequential Dopri5 scan. 4 workgroups x 16 batch rows; 16 waves
// each own a 16-column slice of d. The A-matrix slice is REGISTER-resident
// (8 x v16bf B-fragments per lane, invariant over all 1023 steps); x is
// exchanged per stage via ping-pong LDS tile; V slices streamed with a
// 2-slot async-to-LDS ring.
// ---------------------------------------------------------------------------
__global__ void __launch_bounds__(512)
scan_kernel(const float* __restrict__ tarr, const float* __restrict__ x0,
            const float* __restrict__ A, const float* __restrict__ bvec,
            const float* __restrict__ V, float* __restrict__ xout) {
  extern __shared__ char smem[];
  bf16_t* Xlds = (bf16_t*)smem;                  // [2][16][XKP]
  float*  Vb   = (float*)(Xlds + 2 * 16 * XKP);  // [2][16][VKP]
  const unsigned vb_off = (unsigned)(size_t)(void*)Vb;

  const int tid = threadIdx.x;
  const int lane = tid & 31, wave = tid >> 5;     // 16 waves
  const int h = lane >> 4, nn = lane & 15;
  const int n0 = wave * 16;                       // 16-column slice per wave
  const int col = n0 + nn;
  const int bBase = blockIdx.x * 16;
  const float dt = tarr[1] - tarr[0];

  // Register-resident B-operand fragments of A for this lane's column.
  // Fragment kt element e holds A[kt*32 + h*16 + e][col].
  v16bf Afrag[8];
#pragma unroll
  for (int kt = 0; kt < 8; ++kt) {
#pragma unroll
    for (int e = 0; e < 16; ++e) {
      Afrag[kt][e] = (bf16_t)A[(size_t)(kt * 32 + h * 16 + e) * DDIM + col];
    }
  }

  // Hermite coefficients per stage: u_j = cp*u[t-1] + cc*u[t] + cn*u[t+1]
  const float CJ[6] = {0.f, 0.2f, 0.3f, 0.8f, 8.f / 9.f, 1.f};
  float cpv[6], ccv[6], cnv[6], cc0[6], cn0[6];
#pragma unroll
  for (int j = 0; j < 6; ++j) {
    float s = CJ[j], s2 = s * s, s3 = s2 * s;
    float h00 = 2.f * s3 - 3.f * s2 + 1.f, h10 = s3 - 2.f * s2 + s;
    float h01 = 3.f * s2 - 2.f * s3, h11 = s3 - s2;
    cpv[j] = -h10; ccv[j] = h00 + h10 - h11; cnv[j] = h01 + h11;
    cc0[j] = h00 - h10 - h11; cn0[j] = h01 + h10 + h11;   // t==0: d0 == d1
  }

  const float bias = bvec[col];

  // Load x0 and emit x_traj[:, 0, :].
  v8f xx;
#pragma unroll
  for (int r = 0; r < 8; ++r) {
    int m = r + 8 * h;
    xx[r] = x0[(size_t)(bBase + m) * DDIM + col];
    xout[((size_t)(bBase + m) * T_LEN) * DDIM + col] = xx[r];
  }

  // Preload V[0] -> slot0, V[1] -> slot1.
  async_v(V, 0, bBase, vb_off + 0u * 16 * VKP * 4, tid);
  async_v(V, 1, bBase, vb_off + 1u * 16 * VKP * 4, tid);
  asm volatile("s_wait_asynccnt 0x0" ::: "memory");
  __syncthreads();

  v8f vP, vC, vN;
#pragma unroll
  for (int r = 0; r < 8; ++r) vC[r] = Vb[(r + 8 * h) * VKP + col];
  vP = vC;
  __syncthreads();   // protect slot0 before step-0's async overwrite

  v8f kk[6];

  for (int t = 0; t < T_LEN - 1; ++t) {
    // Prefetch V[t+2] into slot t%2; drain only the previous batch.
    if (t + 2 < T_LEN) {
      async_v(V, t + 2, bBase, vb_off + (unsigned)((t & 1) * 16 * VKP * 4), tid);
      asm volatile("s_wait_asynccnt 0x2" ::: "memory");
    } else {
      asm volatile("s_wait_asynccnt 0x0" ::: "memory");
    }
    __syncthreads();

    if (t > 0) { vP = vC; vC = vN; }
    {
      const float* vs = Vb + ((t + 1) & 1) * 16 * VKP;
#pragma unroll
      for (int r = 0; r < 8; ++r) vN[r] = vs[(r + 8 * h) * VKP + col];
    }

#pragma unroll
    for (int j = 0; j < 6; ++j) {
      v8f xj = xx;
#pragma unroll
      for (int i = 0; i < 5; ++i) {
        if (i < j && AT_[j][i] != 0.f) {
          float c = dt * AT_[j][i];
          xj += c * kk[i];
        }
      }
      // Publish x_j (bf16) to ping-pong tile.
      bf16_t* xb = Xlds + (j & 1) * (16 * XKP);
#pragma unroll
      for (int r = 0; r < 8; ++r) xb[(r + 8 * h) * XKP + col] = (bf16_t)xj[r];
      __syncthreads();

      float fcp = (t == 0) ? 0.f    : cpv[j];
      float fcc = (t == 0) ? cc0[j] : ccv[j];
      float fcn = (t == 0) ? cn0[j] : cnv[j];
      v8f acc = fcp * vP + fcc * vC + fcn * vN + bias;

      const bf16_t* xr = Xlds + (j & 1) * (16 * XKP) + (size_t)(lane & 15) * XKP;
#pragma unroll
      for (int kt = 0; kt < 8; ++kt) {
        int kb = kt * 32;
        v16bf af = frag_lds(xr + kb + h * 16, xr + kb + h * 16 + 8);
        acc = wmma_bf16(af, Afrag[kt], acc);
      }
#pragma unroll
      for (int e = 0; e < 8; ++e) kk[j][e] = tanhf(acc[e]);
    }

    v8f xn = xx;
#pragma unroll
    for (int j = 0; j < 6; ++j) {
      if (BT_[j] != 0.f) {
        float c = dt * BT_[j];
        xn += c * kk[j];
      }
    }
    xx = xn;
#pragma unroll
    for (int r = 0; r < 8; ++r)
      xout[((size_t)(bBase + r + 8 * h) * T_LEN + (t + 1)) * DDIM + col] = xx[r];
  }
}

// ---------------------------------------------------------------------------
// Kernel 3: y = x_traj @ C   (M=65536, N=64, K=256)
// ---------------------------------------------------------------------------
__global__ void __launch_bounds__(256)
y_gemm_kernel(const float* __restrict__ C, const float* __restrict__ X,
              float* __restrict__ Y) {
  extern __shared__ char smem[];
  bf16_t* Ccm = (bf16_t*)smem;   // [64][CKP] column-major bf16
  const int tid = threadIdx.x;
  for (int i = tid; i < DDIM * PDIM; i += 256) {
    int k = i >> 6, n = i & 63;
    Ccm[(size_t)n * CKP + k] = (bf16_t)C[i];
  }
  __syncthreads();
  const int lane = tid & 31, wave = tid >> 5;
  const int h = lane >> 4, nn = lane & 15, mA = lane & 15;
  for (int task = blockIdx.x * 8 + wave; task < (BATCH * T_LEN) / 16; task += 512) {
    const float* xrow = X + (size_t)(task * 16 + mA) * DDIM;
    v8f acc[4] = {};
#pragma unroll
    for (int kt = 0; kt < 8; ++kt) {
      int kb = kt * 32;
      v16bf af = cat16(cvt8(xrow + kb + h * 8), cvt8(xrow + kb + 16 + h * 8));
#pragma unroll
      for (int nt = 0; nt < 4; ++nt) {
        const bf16_t* bc = Ccm + (size_t)(nt * 16 + nn) * CKP + kb + h * 8;
        acc[nt] = wmma_bf16(af, frag_lds(bc, bc + 16), acc[nt]);
      }
    }
#pragma unroll
    for (int nt = 0; nt < 4; ++nt)
#pragma unroll
      for (int r = 0; r < 8; ++r)
        Y[(size_t)(task * 16 + r + 8 * h) * PDIM + nt * 16 + nn] = acc[nt][r];
  }
}

extern "C" void kernel_launch(void* const* d_in, const int* in_sizes, int n_in,
                              void* d_out, int out_size, void* d_ws, size_t ws_size,
                              hipStream_t stream) {
  (void)in_sizes; (void)n_in; (void)out_size; (void)ws_size;
  const float* t  = (const float*)d_in[0];
  const float* u  = (const float*)d_in[1];
  const float* x0 = (const float*)d_in[2];
  const float* A  = (const float*)d_in[3];
  const float* Bm = (const float*)d_in[4];
  const float* b  = (const float*)d_in[5];
  const float* C  = (const float*)d_in[6];
  float* xout = (float*)d_out;                                  // (B,T,d) f32
  float* yout = xout + (size_t)BATCH * T_LEN * DDIM;            // (B,T,p) f32
  float* V = (float*)d_ws;                                      // [T][B][d] f32, 64 MB

  size_t lds1 = (size_t)DDIM * BKP * sizeof(bf16_t);
  v_gemm_kernel<<<64, 256, lds1, stream>>>(u, Bm, V);

  size_t lds2 = (size_t)2 * 16 * XKP * sizeof(bf16_t) +
                (size_t)2 * 16 * VKP * sizeof(float);           // ~50 KB
  scan_kernel<<<4, 512, lds2, stream>>>(t, x0, A, b, V, xout);

  size_t lds3 = (size_t)PDIM * CKP * sizeof(bf16_t);
  y_gemm_kernel<<<64, 256, lds3, stream>>>(C, xout, yout);
}
